// FeatureCorrelation_61521111548046
// MI455X (gfx1250) — matrix operand order repaired and checked by reading, not compiled, not run on hardware
//
#include <hip/hip_runtime.h>

typedef __attribute__((ext_vector_type(16))) _Float16 v16h;
typedef __attribute__((ext_vector_type(8)))  _Float16 v8h;
typedef __attribute__((ext_vector_type(8)))  float    v8f;

static constexpr int BATCH = 64;
static constexpr int CC    = 128;   // channels (K dim of GEMM)
static constexpr int NN    = 1024;  // H*W
static constexpr int NP    = 1025;  // with dustbin
static constexpr int LD    = 1040;  // padded row stride (f16), 16-mult

// ---------------------------------------------------------------------------
// 1) Prep: per (b, n) column of [b, C, n]: subtract mean over C, scale by
//    rsqrt(sum sq), store f16 transposed [b][n][C] (K-contiguous for WMMA).
// ---------------------------------------------------------------------------
__global__ void prep_kernel(const float* __restrict__ A, const float* __restrict__ Bm,
                            _Float16* __restrict__ At, _Float16* __restrict__ Bt) {
  const float*    src = blockIdx.z ? Bm : A;
  _Float16*       dst = blockIdx.z ? Bt : At;
  const int b  = blockIdx.x;
  const int tx = threadIdx.x;           // 0..31 : column within group
  const int ty = threadIdx.y;           // 0..7  : channel group
  const int n  = blockIdx.y * 32 + tx;

  const float* col = src + (size_t)b * CC * NN + n;
  float vals[16];
  float sum = 0.f;
  #pragma unroll
  for (int k = 0; k < 16; ++k) {
    vals[k] = col[(size_t)(ty * 16 + k) * NN];
    sum += vals[k];
  }
  __shared__ float red[8][33];
  red[ty][tx] = sum;
  __syncthreads();
  if (ty == 0) {
    float tot = 0.f;
    #pragma unroll
    for (int q = 0; q < 8; ++q) tot += red[q][tx];
    red[0][tx] = tot;
  }
  __syncthreads();
  const float mean = red[0][tx] * (1.0f / (float)CC);
  __syncthreads();

  float ss = 0.f;
  #pragma unroll
  for (int k = 0; k < 16; ++k) { vals[k] -= mean; ss += vals[k] * vals[k]; }
  red[ty][tx] = ss;
  __syncthreads();
  if (ty == 0) {
    float tot = 0.f;
    #pragma unroll
    for (int q = 0; q < 8; ++q) tot += red[q][tx];
    red[0][tx] = tot;
  }
  __syncthreads();
  const float inv = rsqrtf(red[0][tx] + 1e-12f);

  _Float16* out = dst + ((size_t)b * NN + n) * CC + ty * 16;
  #pragma unroll
  for (int k = 0; k < 16; ++k) out[k] = (_Float16)(vals[k] * inv);
}

// ---------------------------------------------------------------------------
// 2) Batched GEMM via WMMA f16->f32, epilogue K = exp(dot + 2) stored f16.
//    One 16x16 tile per wave; 256 thr = 8 waves/block; 4096 tiles/batch.
// ---------------------------------------------------------------------------
__global__ void pxy_exp_kernel(const _Float16* __restrict__ At,
                               const _Float16* __restrict__ Bt,
                               _Float16* __restrict__ K) {
  const int b    = blockIdx.x;
  const int wave = threadIdx.x >> 5;
  const int lane = threadIdx.x & 31;
  const int tile = blockIdx.y * 8 + wave;          // 0..4095
  const int tm   = (tile >> 6) * 16;
  const int tn   = (tile & 63) * 16;
  const int rc   = lane & 15;                      // A row / B col within tile
  const int half = lane >> 4;                      // lane half selects K slice

  const _Float16* aBase = At + (((size_t)b * NN) + tm + rc) * CC;
  const _Float16* bBase = Bt + (((size_t)b * NN) + tn + rc) * CC;

  v8f acc = {};
  #pragma unroll
  for (int k0 = 0; k0 < CC; k0 += 32) {
    // A 16x32 f16 layout: lanes 0-15 K {0..7,16..23}; lanes 16-31 K {8..15,24..31}
    const _Float16* ap = aBase + k0 + half * 8;
    v8h alo = *(const v8h*)(ap);
    v8h ahi = *(const v8h*)(ap + 16);
    v16h a = __builtin_shufflevector(alo, ahi, 0,1,2,3,4,5,6,7,8,9,10,11,12,13,14,15);
    // B 32x16 f16 layout: lanes 0-15 K {0..7,16..23}; lanes 16-31 K {8..15,24..31} (per column)
    const _Float16* bp = bBase + k0 + half * 8;
    v8h blo = *(const v8h*)(bp);
    v8h bhi = *(const v8h*)(bp + 16);
    v16h bb = __builtin_shufflevector(blo, bhi, 0,1,2,3,4,5,6,7,8,9,10,11,12,13,14,15);
    acc = __builtin_amdgcn_wmma_f32_16x16x32_f16(false, a, false, bb,
                                                 (short)0, acc, false, false);
  }

  // C/D layout: VGPR r -> row (half?8:0)+r, col = lane&15
  const size_t base = (size_t)b * LD * LD;
  const int rbase = tm + (half ? 8 : 0);
  #pragma unroll
  for (int r = 0; r < 8; ++r) {
    float v = __expf(acc[r] + 2.0f);
    K[base + (size_t)(rbase + r) * LD + tn + rc] = (_Float16)v;
  }
}

// ---------------------------------------------------------------------------
// 3) Fill dustbin row/col with exp(alpha), zero pad columns, init r/s.
// ---------------------------------------------------------------------------
__global__ void fill_kernel(const float* __restrict__ bin, _Float16* __restrict__ K,
                            float* __restrict__ r, float* __restrict__ s) {
  const int b = blockIdx.x;
  const int t = blockIdx.y * 256 + threadIdx.x;   // 0..1279
  const float ebin = __expf(*bin);
  const size_t base = (size_t)b * LD * LD;
  if (t < NN) {
    _Float16* p = K + base + (size_t)t * LD + NN;  // cols 1024..1039
    p[0] = (_Float16)ebin;
    #pragma unroll
    for (int q = 1; q < 16; ++q) p[q] = (_Float16)0.f;
  } else if (t == NN) {
    _Float16* p = K + base + (size_t)NN * LD;      // dustbin row
    for (int j = 0; j < LD; ++j) p[j] = (j < NP) ? (_Float16)ebin : (_Float16)0.f;
  }
  if (t < LD) {
    s[(size_t)b * LD + t] = (t < NP) ? 1.0f : 0.0f;  // v=0 -> s=1
    r[(size_t)b * LD + t] = 0.0f;
  }
}

// ---------------------------------------------------------------------------
// 4a) Row phase: r_i = mu_i / sum_j K[i][j] * s_j   (wave per row, s in LDS)
// ---------------------------------------------------------------------------
__global__ void row_phase(const _Float16* __restrict__ K,
                          const float* __restrict__ s, float* __restrict__ r) {
  __shared__ float s_sh[LD];
  const int b = blockIdx.x;
  for (int idx = threadIdx.x; idx < LD; idx += 256)
    s_sh[idx] = s[(size_t)b * LD + idx];
  __syncthreads();

  const int wave = threadIdx.x >> 5;
  const int lane = threadIdx.x & 31;
  const int i = blockIdx.y * 8 + wave;
  if (i >= NP) return;

  const _Float16* row = K + ((size_t)b * LD + i) * LD;
  float sum = 0.f;
  #pragma unroll
  for (int c = 0; c < 5; ++c) {
    const int j = (c * 32 + lane) * 8;
    if (j < LD) {
      v8h kv = *(const v8h*)(row + j);
      #pragma unroll
      for (int e = 0; e < 8; ++e) sum += (float)kv[e] * s_sh[j + e];
    }
  }
  #pragma unroll
  for (int off = 16; off > 0; off >>= 1) sum += __shfl_xor(sum, off, 32);
  if (lane == 0) {
    const float mu = (i < NN) ? (1.0f / 2048.0f) : 0.5f;
    r[(size_t)b * LD + i] = mu / sum;
  }
}

// ---------------------------------------------------------------------------
// 4b) Col phase: s_j = nu_j / sum_i K[i][j] * r_i   (thread per column)
// ---------------------------------------------------------------------------
__global__ void col_phase(const _Float16* __restrict__ K,
                          const float* __restrict__ r, float* __restrict__ s) {
  const int b = blockIdx.x;
  const int j = blockIdx.y * 256 + threadIdx.x;   // 0..1279
  __shared__ float r_sh[256];
  const _Float16* Kb = K + (size_t)b * LD * LD;
  float sum = 0.f;
  for (int i0 = 0; i0 < NP; i0 += 256) {
    __syncthreads();
    const int ii = i0 + threadIdx.x;
    r_sh[threadIdx.x] = (ii < NP) ? r[(size_t)b * LD + ii] : 0.f;
    __syncthreads();
    const int lim = (NP - i0 < 256) ? (NP - i0) : 256;
    if (j < LD) {
      for (int k = 0; k < lim; ++k)
        sum += (float)Kb[(size_t)(i0 + k) * LD + j] * r_sh[k];
    }
  }
  if (j < NP) {
    const float nu = (j < NN) ? (1.0f / 2048.0f) : 0.5f;
    s[(size_t)b * LD + j] = nu / sum;
  }
}

// ---------------------------------------------------------------------------
// 5) ot_b = rsqrt(2048 * sum_{i,j<1024} K r s log(K)); two-stage deterministic reduce
// ---------------------------------------------------------------------------
__global__ void final_partial(const _Float16* __restrict__ K,
                              const float* __restrict__ r, const float* __restrict__ s,
                              float* __restrict__ partial) {
  const int b = blockIdx.x;
  __shared__ float s_sh[NN];
  for (int idx = threadIdx.x; idx < NN; idx += 256)
    s_sh[idx] = s[(size_t)b * LD + idx];
  __syncthreads();

  float acc = 0.f;
  for (int rr = 0; rr < 32; ++rr) {
    const int i = blockIdx.y * 32 + rr;
    const float ri = r[(size_t)b * LD + i];
    const _Float16* row = K + ((size_t)b * LD + i) * LD;
    for (int j = threadIdx.x; j < NN; j += 256) {
      const float k = (float)row[j];
      acc += ri * k * s_sh[j] * __logf(k);   // pxy = log(K)
    }
  }
  __shared__ float red[256];
  red[threadIdx.x] = acc;
  __syncthreads();
  for (int off = 128; off > 0; off >>= 1) {
    if (threadIdx.x < off) red[threadIdx.x] += red[threadIdx.x + off];
    __syncthreads();
  }
  if (threadIdx.x == 0) partial[b * 32 + blockIdx.y] = red[0];
}

__global__ void final_reduce(const float* __restrict__ partial, float* __restrict__ out) {
  const int b = blockIdx.x;
  float v = partial[b * 32 + threadIdx.x];
  #pragma unroll
  for (int off = 16; off > 0; off >>= 1) v += __shfl_xor(v, off, 32);
  if (threadIdx.x == 0) out[b] = rsqrtf(2048.0f * v);
}

// ---------------------------------------------------------------------------
extern "C" void kernel_launch(void* const* d_in, const int* in_sizes, int n_in,
                              void* d_out, int out_size, void* d_ws, size_t ws_size,
                              hipStream_t stream) {
  const float* fA  = (const float*)d_in[0];
  const float* fB  = (const float*)d_in[1];
  const float* bin = (const float*)d_in[2];
  float* out = (float*)d_out;

  char* ws = (char*)d_ws;
  constexpr size_t OFF_AT = 0;
  constexpr size_t SZ_T   = (size_t)BATCH * NN * CC * 2;          // 16.8 MB
  constexpr size_t OFF_BT = OFF_AT + SZ_T;
  constexpr size_t OFF_K  = OFF_BT + SZ_T;
  constexpr size_t SZ_K   = (size_t)BATCH * LD * LD * 2;          // 138 MB (L2-resident)
  constexpr size_t OFF_R  = OFF_K + SZ_K;
  constexpr size_t SZ_V   = (size_t)BATCH * LD * 4;
  constexpr size_t OFF_S  = OFF_R + SZ_V;
  constexpr size_t OFF_P  = OFF_S + SZ_V;

  _Float16* At = (_Float16*)(ws + OFF_AT);
  _Float16* Bt = (_Float16*)(ws + OFF_BT);
  _Float16* K  = (_Float16*)(ws + OFF_K);
  float* r = (float*)(ws + OFF_R);
  float* s = (float*)(ws + OFF_S);
  float* partial = (float*)(ws + OFF_P);

  prep_kernel<<<dim3(BATCH, NN / 32, 2), dim3(32, 8), 0, stream>>>(fA, fB, At, Bt);
  pxy_exp_kernel<<<dim3(BATCH, 512), 256, 0, stream>>>(At, Bt, K);
  fill_kernel<<<dim3(BATCH, 5), 256, 0, stream>>>(bin, K, r, s);

  for (int it = 0; it < 100; ++it) {
    row_phase<<<dim3(BATCH, 129), 256, 0, stream>>>(K, s, r);
    col_phase<<<dim3(BATCH, 5), 256, 0, stream>>>(K, r, s);
  }

  final_partial<<<dim3(BATCH, 32), 256, 0, stream>>>(K, r, s, partial);
  final_reduce<<<BATCH, 32, 0, stream>>>(partial, out);
}
